// Devign2_32693291057853
// MI455X (gfx1250) — matrix operand order repaired and checked by reading, not compiled, not run on hardware
//
#include <hip/hip_runtime.h>
#include <hip/hip_bf16.h>

// ---------------------------------------------------------------------------
// Devign-style GNN pipeline for MI455X (gfx1250, wave32, WMMA).
//   encoder GEMM -> [6x GGNN layer: msg GEMM, edge scatter, 2x GRU GEMM,
//   GRU elementwise] -> mean pool -> 3-layer MLP -> sigmoid.
// GEMMs: v_wmma_f32_16x16x32_bf16, fp32 accumulate.
//  - staging is branch-free: clamped vector loads + cndmask zero-fill
//    (K is a multiple of 8 at every call site, so 8-wide chunks are never
//    partially in-bounds)
//  - LDS is double-buffered: one barrier per K-step, loads overlap WMMA
// ---------------------------------------------------------------------------

#define NN    100000
#define EE    1600000
#define GG    256
#define D_IN  768
#define D_C   101
#define HH    200
#define LL    6

typedef __attribute__((ext_vector_type(16))) __bf16 v16bf;
typedef __attribute__((ext_vector_type(8)))  __bf16 v8bf;
typedef __attribute__((ext_vector_type(8)))  float  v8f;

#define BM 64
#define BN 128
#define BK 32

// ---- branch-free tile loaders: global fp32 -> bf16 registers -------------
// Requires K % 8 == 0 and M >= 1, Nc >= 1, K >= 8 (true for all call sites).
__device__ __forceinline__ void load_tileA(const float* __restrict__ A, int lda,
                                           int M, int K, int blockRow,
                                           int ar, int ac0, int k0, v8bf& tA) {
    int grow = blockRow + ar;
    const bool rowOK = grow < M;
    if (!rowOK) grow = M - 1;                 // clamp: stay in allocation
    const int kbase = k0 + ac0;               // multiple of 8
    const bool kOK = (kbase + 8) <= K;        // chunk fully in or fully out
    const int koff = kOK ? kbase : 0;         // clamped, still 32B aligned
    const float* ap = A + (size_t)grow * lda + koff;
    const float4 v0 = *(const float4*)ap;
    const float4 v1 = *(const float4*)(ap + 4);
    const float vv[8] = {v0.x, v0.y, v0.z, v0.w, v1.x, v1.y, v1.z, v1.w};
    const bool ok = rowOK && kOK;
    #pragma unroll
    for (int kk = 0; kk < 8; ++kk)
        tA[kk] = ok ? (__bf16)vv[kk] : (__bf16)0.0f;
}

__device__ __forceinline__ void load_tileB(const float* __restrict__ B, int ldb,
                                           int transB, int Nc, int K,
                                           int blockCol, int bn, int bk0, int k0,
                                           v8bf& tB0, v8bf& tB1) {
    int gn = blockCol + bn;
    const bool nOK = gn < Nc;
    if (!nOK) gn = Nc - 1;                    // clamp
    const int kb0 = k0 + bk0;                 // multiple of 16 (hence of 8)
    const int kb1 = kb0 + 8;
    const bool k0OK = (kb0 + 8) <= K;
    const bool k1OK = (kb1 + 8) <= K;
    if (transB) {
        const int koff0 = k0OK ? kb0 : 0;
        const int koff1 = k1OK ? kb1 : 0;
        const float* bp = B + (size_t)gn * ldb;
        const float4 v0 = *(const float4*)(bp + koff0);
        const float4 v1 = *(const float4*)(bp + koff0 + 4);
        const float4 v2 = *(const float4*)(bp + koff1);
        const float4 v3 = *(const float4*)(bp + koff1 + 4);
        const float a[8] = {v0.x, v0.y, v0.z, v0.w, v1.x, v1.y, v1.z, v1.w};
        const float b[8] = {v2.x, v2.y, v2.z, v2.w, v3.x, v3.y, v3.z, v3.w};
        const bool ok0 = nOK && k0OK, ok1 = nOK && k1OK;
        #pragma unroll
        for (int kk = 0; kk < 8; ++kk) {
            tB0[kk] = ok0 ? (__bf16)a[kk] : (__bf16)0.0f;
            tB1[kk] = ok1 ? (__bf16)b[kk] : (__bf16)0.0f;
        }
    } else {
        // strided gather over k (weights are tiny, L2-resident)
        const bool ok0 = nOK && k0OK, ok1 = nOK && k1OK;
        const int koff0 = k0OK ? kb0 : 0;
        const int koff1 = k1OK ? kb1 : 0;
        const float* bp0 = B + (size_t)koff0 * ldb + gn;
        const float* bp1 = B + (size_t)koff1 * ldb + gn;
        #pragma unroll
        for (int kk = 0; kk < 8; ++kk) {
            const float f0 = bp0[(size_t)kk * ldb];
            const float f1 = bp1[(size_t)kk * ldb];
            tB0[kk] = ok0 ? (__bf16)f0 : (__bf16)0.0f;
            tB1[kk] = ok1 ? (__bf16)f1 : (__bf16)0.0f;
        }
    }
}

// ---------------------------------------------------------------------------
// Tiled bf16-WMMA GEMM:  C[M,Nc] = A[M,K](f32,row-major) * B + bias
//   transB==1: B[k][n] = Bsrc[n*ldb + k]
//   transB==0: B[k][n] = Bsrc[k*ldb + n]
// EPI==0: plain store (+bias). EPI==1: encoder (compressed + padded h).
// ---------------------------------------------------------------------------
template <int EPI>
__global__ __launch_bounds__(256)
void gemm_bf16_wmma(const float* __restrict__ A, int lda,
                    const float* __restrict__ Bsrc, int ldb, int transB,
                    const float* __restrict__ bias,
                    float* __restrict__ C, int ldc,
                    float* __restrict__ C2, int ldc2,
                    int M, int Nc, int K) {
    __shared__ __align__(16) __bf16 As[2][BM][BK + 8];   // ping-pong
    __shared__ __align__(16) __bf16 Bt[2][BN][BK + 8];

    const int t = threadIdx.x;
    const int blockRow = blockIdx.y * BM;
    const int blockCol = blockIdx.x * BN;

    const int lane = t & 31;
    const int wave = t >> 5;
    const int l16  = lane & 15;
    const int lh   = lane >> 4;
    const int wr   = (wave >> 2) * 32;
    const int wc   = (wave & 3) * 32;

    v8f acc[2][2];
    #pragma unroll
    for (int i = 0; i < 2; ++i)
        #pragma unroll
        for (int j = 0; j < 2; ++j)
            #pragma unroll
            for (int p = 0; p < 8; ++p) acc[i][j][p] = 0.0f;

    // staging index maps
    const int ar  = t >> 2;          // 0..63
    const int ac0 = (t & 3) * 8;     // 0/8/16/24
    const int bn  = t >> 1;          // 0..127
    const int bk0 = (t & 1) * 16;    // 0/16

    // prologue: tile 0 -> regs -> LDS buf 0
    v8bf tA, tB0, tB1;
    load_tileA(A, lda, M, K, blockRow, ar, ac0, 0, tA);
    load_tileB(Bsrc, ldb, transB, Nc, K, blockCol, bn, bk0, 0, tB0, tB1);
    *(v8bf*)&As[0][ar][ac0]     = tA;
    *(v8bf*)&Bt[0][bn][bk0]     = tB0;
    *(v8bf*)&Bt[0][bn][bk0 + 8] = tB1;
    __syncthreads();

    int buf = 0;
    for (int k0 = 0; k0 < K; k0 += BK) {
        const bool hasNext = (k0 + BK) < K;
        // issue next tile's global loads (overlap with WMMA below)
        if (hasNext) {
            load_tileA(A, lda, M, K, blockRow, ar, ac0, k0 + BK, tA);
            load_tileB(Bsrc, ldb, transB, Nc, K, blockCol, bn, bk0, k0 + BK,
                       tB0, tB1);
        }

        // ---- fragments (ds_load_b128 pairs) + WMMA from As/Bt[buf] ----
        // A 16x32 bf16: lanes<16: M=lane, K {0..7,16..23}; lanes>=16: {8..15,24..31}
        const int kb0 = lh * 8;
        v16bf afrag[2], bfrag[2];
        #pragma unroll
        for (int i = 0; i < 2; ++i) {
            const int am = wr + i * 16 + l16;
            const v8bf lo = *(const v8bf*)&As[buf][am][kb0];
            const v8bf hi = *(const v8bf*)&As[buf][am][kb0 + 16];
            afrag[i] = __builtin_shufflevector(lo, hi, 0, 1, 2, 3, 4, 5, 6, 7,
                                               8, 9, 10, 11, 12, 13, 14, 15);
        }
        // B 32x16 bf16: lanes<16: N=lane, K {0..7,16..23}; lanes>=16: {8..15,24..31}
        const int kbB = lh * 8;
        #pragma unroll
        for (int j = 0; j < 2; ++j) {
            const int bnl = wc + j * 16 + l16;
            const v8bf lo = *(const v8bf*)&Bt[buf][bnl][kbB];
            const v8bf hi = *(const v8bf*)&Bt[buf][bnl][kbB + 16];
            bfrag[j] = __builtin_shufflevector(lo, hi, 0, 1, 2, 3, 4, 5, 6, 7,
                                               8, 9, 10, 11, 12, 13, 14, 15);
        }
        #pragma unroll
        for (int i = 0; i < 2; ++i)
            #pragma unroll
            for (int j = 0; j < 2; ++j)
                acc[i][j] = __builtin_amdgcn_wmma_f32_16x16x32_bf16(
                    false, afrag[i], false, bfrag[j],
                    (short)0, acc[i][j], false, false);

        // commit next tile to the other buffer; single barrier per K-step
        if (hasNext) {
            const int nb = buf ^ 1;
            *(v8bf*)&As[nb][ar][ac0]     = tA;
            *(v8bf*)&Bt[nb][bn][bk0]     = tB0;
            *(v8bf*)&Bt[nb][bn][bk0 + 8] = tB1;
            __syncthreads();
            buf = nb;
        }
    }

    // ---- epilogue (C 16x16 f32 layout: VGPR p -> M = p + 8*lh, N = l16) ----
    #pragma unroll
    for (int i = 0; i < 2; ++i) {
        #pragma unroll
        for (int j = 0; j < 2; ++j) {
            const int rbase = blockRow + wr + i * 16 + lh * 8;
            const int c     = blockCol + wc + j * 16 + l16;
            #pragma unroll
            for (int p = 0; p < 8; ++p) {
                const int r = rbase + p;
                if (r < M && c < Nc) {
                    float v = acc[i][j][p];
                    if (bias) v += bias[c];
                    if (EPI == 0) {
                        C[(size_t)r * ldc + c] = v;
                    } else {
                        C[(size_t)r * ldc + c]   = v;   // compressed [N,101]
                        C2[(size_t)r * ldc2 + c] = v;   // h padded   [N,200]
                    }
                }
            }
        }
    }
}

// ---------------------------------------------------------------------------
__global__ void fill_zero_kernel(float* __restrict__ p, long long n) {
    long long i = (long long)blockIdx.x * blockDim.x + threadIdx.x;
    if (i < n) p[i] = 0.0f;
}

// Edge scatter: agg[dst] += m[src]; 8 floats/thread, native fp32 atomics.
__global__ void scatter_add_kernel(const float* __restrict__ m,
                                   const int* __restrict__ ei,
                                   float* __restrict__ agg) {
    const int idx = blockIdx.x * blockDim.x + threadIdx.x;   // E*25 threads
    if (idx >= EE * 25) return;
    const int e    = idx / 25;
    const int part = idx - e * 25;
    const int s = ei[e];
    const int d = ei[EE + e];
    const float* __restrict__ sp = m + (size_t)s * HH + part * 8;
    float* __restrict__ dp = agg + (size_t)d * HH + part * 8;
    const float4 a0 = *(const float4*)sp;       // 32B coalesced gather
    const float4 a1 = *(const float4*)(sp + 4);
    const float vv[8] = {a0.x, a0.y, a0.z, a0.w, a1.x, a1.y, a1.z, a1.w};
    #pragma unroll
    for (int j = 0; j < 8; ++j)
        __hip_atomic_fetch_add(&dp[j], vv[j], __ATOMIC_RELAXED,
                               __HIP_MEMORY_SCOPE_AGENT);
}

// GRU cell elementwise: h = (1-z)*n + z*h
__global__ void gru_elem_kernel(const float* __restrict__ gi,
                                const float* __restrict__ gh,
                                float* __restrict__ h) {
    const int idx = blockIdx.x * blockDim.x + threadIdx.x;   // N*H
    if (idx >= NN * HH) return;
    const int n = idx / HH;
    const int f = idx - n * HH;
    const float* gin = gi + (size_t)n * (3 * HH);
    const float* ghn = gh + (size_t)n * (3 * HH);
    const float i_r = gin[f], i_z = gin[HH + f], i_n = gin[2 * HH + f];
    const float h_r = ghn[f], h_z = ghn[HH + f], h_n = ghn[2 * HH + f];
    const float r = 1.0f / (1.0f + __expf(-(i_r + h_r)));
    const float z = 1.0f / (1.0f + __expf(-(i_z + h_z)));
    const float nn = tanhf(i_n + r * h_n);
    h[idx] = (1.0f - z) * nn + z * h[idx];
}

// Mean-pool accumulate: feat = concat(h, compressed), 301 cols
__global__ void pool_accum_kernel(const float* __restrict__ h,
                                  const float* __restrict__ comp,
                                  const int* __restrict__ batch,
                                  float* __restrict__ sums,
                                  float* __restrict__ cnt) {
    const int FC = HH + D_C;  // 301
    const int idx = blockIdx.x * blockDim.x + threadIdx.x;
    if (idx >= NN * FC) return;
    const int n = idx / FC;
    const int c = idx - n * FC;
    const int g = batch[n];
    const float v = (c < HH) ? h[(size_t)n * HH + c]
                             : comp[(size_t)n * D_C + (c - HH)];
    __hip_atomic_fetch_add(&sums[(size_t)g * FC + c], v, __ATOMIC_RELAXED,
                           __HIP_MEMORY_SCOPE_AGENT);
    if (c == 0)
        __hip_atomic_fetch_add(&cnt[g], 1.0f, __ATOMIC_RELAXED,
                               __HIP_MEMORY_SCOPE_AGENT);
}

__global__ void pool_final_kernel(const float* __restrict__ sums,
                                  const float* __restrict__ cnt,
                                  float* __restrict__ graph) {
    const int FC = HH + D_C;
    const int idx = blockIdx.x * blockDim.x + threadIdx.x;
    if (idx >= GG * FC) return;
    const int g = idx / FC;
    graph[idx] = sums[idx] / fmaxf(cnt[g], 1.0f);
}

// Classifier: 301 -> 256 relu -> 128 relu -> 1 sigmoid; one block per graph.
__global__ __launch_bounds__(256)
void classifier_kernel(const float* __restrict__ graph,
                       const float* __restrict__ W1, const float* __restrict__ b1,
                       const float* __restrict__ W2, const float* __restrict__ b2,
                       const float* __restrict__ W3, const float* __restrict__ b3,
                       float* __restrict__ out) {
    const int FC = HH + D_C;
    __shared__ float feat[FC];
    __shared__ float z1[256];
    __shared__ float z2[128];
    const int g = blockIdx.x;
    const int t = threadIdx.x;
    for (int c = t; c < FC; c += 256) feat[c] = graph[(size_t)g * FC + c];
    __syncthreads();
    float s = b1[t];
    for (int k = 0; k < FC; ++k) s = fmaf(feat[k], W1[(size_t)t * FC + k], s);
    z1[t] = fmaxf(s, 0.0f);
    __syncthreads();
    if (t < 128) {
        float s2 = b2[t];
        for (int k = 0; k < 256; ++k) s2 = fmaf(z1[k], W2[(size_t)t * 256 + k], s2);
        z2[t] = fmaxf(s2, 0.0f);
    }
    __syncthreads();
    if (t == 0) {
        float s3 = b3[0];
        for (int k = 0; k < 128; ++k) s3 = fmaf(z2[k], W3[k], s3);
        out[g] = 1.0f / (1.0f + __expf(-s3));
    }
}

// ---------------------------------------------------------------------------
static inline size_t align256(size_t x) { return (x + 255) & ~(size_t)255; }

extern "C" void kernel_launch(void* const* d_in, const int* in_sizes, int n_in,
                              void* d_out, int out_size, void* d_ws, size_t ws_size,
                              hipStream_t stream) {
    (void)in_sizes; (void)n_in; (void)out_size; (void)ws_size;

    const float* x       = (const float*)d_in[0];
    const int*   ei      = (const int*)d_in[1];
    const int*   batch   = (const int*)d_in[2];
    const float* enc_W   = (const float*)d_in[3];
    const float* enc_b   = (const float*)d_in[4];
    const float* ggnn_W  = (const float*)d_in[5];
    const float* W_ih    = (const float*)d_in[6];
    const float* b_ih    = (const float*)d_in[7];
    const float* W_hh    = (const float*)d_in[8];
    const float* b_hh    = (const float*)d_in[9];
    const float* W1      = (const float*)d_in[10];
    const float* b1      = (const float*)d_in[11];
    const float* W2      = (const float*)d_in[12];
    const float* b2      = (const float*)d_in[13];
    const float* W3      = (const float*)d_in[14];
    const float* b3      = (const float*)d_in[15];
    float* out           = (float*)d_out;

    const int FC = HH + D_C;  // 301

    // workspace carve-out
    char* w = (char*)d_ws;
    float* h     = (float*)w; w += align256((size_t)NN * HH * 4);
    float* comp  = (float*)w; w += align256((size_t)NN * D_C * 4);
    float* m     = (float*)w; w += align256((size_t)NN * HH * 4);
    float* agg   = (float*)w; w += align256((size_t)NN * HH * 4);
    float* gi    = (float*)w; w += align256((size_t)NN * 3 * HH * 4);
    float* gh    = (float*)w; w += align256((size_t)NN * 3 * HH * 4);
    float* sums  = (float*)w; w += align256((size_t)GG * FC * 4);
    float* cnt   = (float*)w; w += align256((size_t)GG * 4);
    float* graph = (float*)w; w += align256((size_t)GG * FC * 4);

    const int nby = (NN + BM - 1) / BM;   // 1563 row-blocks

    // 1) zero h (encoder only writes cols < 101)
    {
        long long n = (long long)NN * HH;
        fill_zero_kernel<<<(int)((n + 255) / 256), 256, 0, stream>>>(h, n);
    }

    // 2) encoder: compressed = x @ enc_W.T + enc_b ; h = pad(compressed)
    gemm_bf16_wmma<1><<<dim3(1, nby), 256, 0, stream>>>(
        x, D_IN, enc_W, D_IN, /*transB=*/1, enc_b,
        comp, D_C, h, HH, NN, D_C, D_IN);

    // 3) GGNN layers
    for (int l = 0; l < LL; ++l) {
        // m = h @ ggnn_weight[l]   (weight [H,H] row-major, no transpose)
        gemm_bf16_wmma<0><<<dim3((HH + BN - 1) / BN, nby), 256, 0, stream>>>(
            h, HH, ggnn_W + (size_t)l * HH * HH, HH, /*transB=*/0, nullptr,
            m, HH, nullptr, 0, NN, HH, HH);

        // agg = scatter_add(m[src] -> dst)
        {
            long long n = (long long)NN * HH;
            fill_zero_kernel<<<(int)((n + 255) / 256), 256, 0, stream>>>(agg, n);
        }
        scatter_add_kernel<<<(EE * 25 + 255) / 256, 256, 0, stream>>>(m, ei, agg);

        // gi = agg @ W_ih.T + b_ih ; gh = h @ W_hh.T + b_hh   ([N, 600])
        gemm_bf16_wmma<0><<<dim3((3 * HH + BN - 1) / BN, nby), 256, 0, stream>>>(
            agg, HH, W_ih, HH, /*transB=*/1, b_ih,
            gi, 3 * HH, nullptr, 0, NN, 3 * HH, HH);
        gemm_bf16_wmma<0><<<dim3((3 * HH + BN - 1) / BN, nby), 256, 0, stream>>>(
            h, HH, W_hh, HH, /*transB=*/1, b_hh,
            gh, 3 * HH, nullptr, 0, NN, 3 * HH, HH);

        // h = GRU(gi, gh, h)
        gru_elem_kernel<<<(NN * HH + 255) / 256, 256, 0, stream>>>(gi, gh, h);
    }

    // 4) mean pool over batch ids
    {
        long long n = (long long)GG * FC;
        fill_zero_kernel<<<(int)((n + 255) / 256), 256, 0, stream>>>(sums, n);
        fill_zero_kernel<<<1, 256, 0, stream>>>(cnt, GG);
    }
    pool_accum_kernel<<<(NN * FC + 255) / 256, 256, 0, stream>>>(h, comp, batch,
                                                                 sums, cnt);
    pool_final_kernel<<<(GG * FC + 255) / 256, 256, 0, stream>>>(sums, cnt, graph);

    // 5) classifier -> sigmoid logits [G,1]
    classifier_kernel<<<GG, 256, 0, stream>>>(graph, W1, b1, W2, b2, W3, b3, out);
}
